// HiLoAttention_11192684773830
// MI455X (gfx1250) — matrix operand order repaired and checked by reading, not compiled, not run on hardware
//
#include <hip/hip_runtime.h>
#include <hip/hip_bf16.h>

typedef __attribute__((ext_vector_type(16))) _Float16 v16h;
typedef __attribute__((ext_vector_type(8)))  _Float16 v8h;
typedef __attribute__((ext_vector_type(8)))  float    v8f;

#define BB 4
#define CC 128
#define NN 4096
#define EE 64
#define NSPLIT 4
#define SPLEN (NN / NSPLIT)          // 1024 keys per split wave
#define PARTSZ 1056                  // 16 m + 16 l + 64*16 O floats per partial
#define INV63 (1.0f/63.0f)
#define BIASK 200.0f                 // 1/(2*0.05^2)

__device__ __forceinline__ v8f wmma16(v16h a, v16h b, v8f c) {
  // v_wmma_f32_16x16x32_f16: D = A(16x32 f16) * B(32x16 f16) + C(16x16 f32)
  return __builtin_amdgcn_wmma_f32_16x16x32_f16(false, a, false, b, (short)0, c, false, false);
}

// ---------------------------------------------------------------------------
// Kernel 1: Q/K/V projections. One wave = 16 pixels of one batch.
// A-frag (16x32 f16): lane l (ll=l&15, lh=l>>4): row M=ll,
//   element i -> K = ch*32 + lh*16 + i
// B-frag (32x16 f16): lane l: col N=ll, element i -> K = ch*32 + lh*16 + i
// All 4 B-fragments are materialized before the WMMA chain so the compiler
// does not have to insert WMMA->VALU WAR hazard NOPs on fragment reuse.
// ---------------------------------------------------------------------------
template<int MODE>
__device__ __forceinline__ void proj_store(v8f acc, int b, int n0, int lh, int ll,
                                           int t, _Float16* __restrict__ dst) {
#pragma unroll
  for (int v = 0; v < 8; ++v) {
    const int nn = n0 + v + 8*lh;           // C-layout row
    const int e  = t*16 + ll;               // C-layout col
    if (MODE == 0) dst[((size_t)b*NN + nn)*EE + e] = (_Float16)(acc[v] * 0.125f); // Q * E^-0.5
    if (MODE == 1) dst[((size_t)b*NN + nn)*EE + e] = (_Float16)acc[v];            // K row-major
    if (MODE == 2) dst[((size_t)b*EE + e)*NN + nn] = (_Float16)acc[v];            // V transposed
  }
}

template<int MODE>
__device__ __forceinline__ void proj_mat(const v16h* Xa, const float* __restrict__ W,
                                         int b, int n0, int lh, int ll,
                                         _Float16* __restrict__ dst) {
#pragma unroll
  for (int t = 0; t < 4; ++t) {             // 4 e-tiles of 16
    v16h Bf[4];
#pragma unroll
    for (int ch = 0; ch < 4; ++ch) {        // load+convert ALL fragments first
      const float* wr = W + (size_t)(t*16 + ll)*CC + ch*32 + lh*16; // W[e][c] contig in c
#pragma unroll
      for (int i = 0; i < 16; ++i) Bf[ch][i] = (_Float16)wr[i];
    }
    v8f acc = {};
#pragma unroll
    for (int ch = 0; ch < 4; ++ch)          // K = C = 128 -> 4 chained WMMAs
      acc = wmma16(Xa[ch], Bf[ch], acc);
    proj_store<MODE>(acc, b, n0, lh, ll, t, dst);
  }
}

__global__ __launch_bounds__(32) void k_proj(const float* __restrict__ feat,
    const float* __restrict__ Wq, const float* __restrict__ Wk,
    const float* __restrict__ Wv, _Float16* __restrict__ Qh,
    _Float16* __restrict__ Kh, _Float16* __restrict__ Vth) {
  const int blk = blockIdx.x, b = blk >> 8, n0 = (blk & 255) << 4;
  const int l = threadIdx.x, lh = l >> 4, ll = l & 15;
  const int n = n0 + ll;
  const float* fb = feat + (size_t)b*CC*NN;

  v16h Xa[4];                                // xf tile 16x128 as 4 A-fragments
#pragma unroll
  for (int ch = 0; ch < 4; ++ch)
#pragma unroll
    for (int i = 0; i < 16; ++i) {
      const int c = ch*32 + lh*16 + i;
      Xa[ch][i] = (_Float16)fb[(size_t)c*NN + n];   // lanes 0..15 contiguous in n
    }

  proj_mat<0>(Xa, Wq, b, n0, lh, ll, Qh);
  proj_mat<1>(Xa, Wk, b, n0, lh, ll, Kh);
  proj_mat<2>(Xa, Wv, b, n0, lh, ll, Vth);
}

// ---------------------------------------------------------------------------
// Kernel 2: split-K flash attention. One wave = 16 queries x 1024 keys.
// Emits unnormalized partial (m, l, O) in f32; O stored e-major [e][row].
// ---------------------------------------------------------------------------
__global__ __launch_bounds__(32) void k_attn(const _Float16* __restrict__ Qh,
    const _Float16* __restrict__ Kh, const _Float16* __restrict__ Vth,
    const unsigned char* __restrict__ maskb, float* __restrict__ part) {
  __shared__ __align__(16) _Float16 Pl[16][48];   // 16x32 P tile, 96B row stride
  const int blk = blockIdx.x;
  const int sp  = blk >> 10;                      // key split 0..3
  const int qblk = blk & 1023;
  const int b = qblk >> 8, n0 = (qblk & 255) << 4;
  const int l = threadIdx.x, lh = l >> 4, ll = l & 15;

  v16h Qa[2];
  {
    const _Float16* qrow = Qh + ((size_t)b*NN + n0 + ll)*EE;
#pragma unroll
    for (int ch = 0; ch < 2; ++ch) {
      v8h lo = *(const v8h*)(qrow + ch*32 + lh*16);
      v8h hi = *(const v8h*)(qrow + ch*32 + lh*16 + 8);
#pragma unroll
      for (int i = 0; i < 8; ++i) { Qa[ch][i] = lo[i]; Qa[ch][8+i] = hi[i]; }
    }
  }

  const float yq = (float)(n0 >> 6) * INV63;      // 16 consecutive n share image row
  float xq[8];
#pragma unroll
  for (int v = 0; v < 8; ++v) xq[v] = (float)((n0 & 63) + v + 8*lh) * INV63;

  float m[8], lsum[8];
  v8f O[4] = {};
#pragma unroll
  for (int v = 0; v < 8; ++v) { m[v] = -1e30f; lsum[v] = 0.0f; }

  const _Float16* kb = Kh  + (size_t)b*NN*EE;
  const _Float16* vb = Vth + (size_t)b*EE*NN;
  const unsigned char* mb = maskb + (size_t)b*NN;

  const int kend = sp*SPLEN + SPLEN;
  for (int k0 = sp*SPLEN; k0 < kend; k0 += 32) {
    const int kA = k0 + ll, kB = kA + 16;
    const _Float16* ra = kb + (size_t)kA*EE + lh*16;
    const _Float16* rb = kb + (size_t)kB*EE + lh*16;
    v16h KA0 = *(const v16h*)(ra);          // B-frag, e 0..31
    v16h KA1 = *(const v16h*)(ra + 32);     // B-frag, e 32..63
    v16h KB0 = *(const v16h*)(rb);
    v16h KB1 = *(const v16h*)(rb + 32);

    v8f z = {};
    v8f S0 = wmma16(Qa[0], KA0, z); S0 = wmma16(Qa[1], KA1, S0);
    v8f S1 = wmma16(Qa[0], KB0, z); S1 = wmma16(Qa[1], KB1, S1);

    // pos bias + hi-key mask (keep lo keys; hi -> sentinel below running-max floor)
    const bool hiA = mb[kA] != 0, hiB = mb[kB] != 0;
    const float ykA = (float)(kA >> 6) * INV63, xkA = (float)(kA & 63) * INV63;
    const float ykB = (float)(kB >> 6) * INV63, xkB = (float)(kB & 63) * INV63;
    const float dyA = yq - ykA, dyB = yq - ykB;
    const float cA = dyA*dyA,  cB = dyB*dyB;
#pragma unroll
    for (int v = 0; v < 8; ++v) {
      const float dxA = xq[v] - xkA, dxB = xq[v] - xkB;
      S0[v] = hiA ? -3e38f : (S0[v] - BIASK*(cA + dxA*dxA));
      S1[v] = hiB ? -3e38f : (S1[v] - BIASK*(cB + dxB*dxB));
    }

    // online softmax: row reductions across the 16-lane half (C-layout N striping)
    float corr[8];
#pragma unroll
    for (int v = 0; v < 8; ++v) {
      float t = fmaxf(S0[v], S1[v]);
      t = fmaxf(t, __shfl_xor(t, 1, 32));
      t = fmaxf(t, __shfl_xor(t, 2, 32));
      t = fmaxf(t, __shfl_xor(t, 4, 32));
      t = fmaxf(t, __shfl_xor(t, 8, 32));
      const float mn = fmaxf(m[v], t);       // floor at -1e30: all-masked block -> P==0
      corr[v] = __expf(m[v] - mn);
      m[v] = mn;
      const float p0 = __expf(S0[v] - mn);
      const float p1 = __expf(S1[v] - mn);
      float s = p0 + p1;
      s += __shfl_xor(s, 1, 32);
      s += __shfl_xor(s, 2, 32);
      s += __shfl_xor(s, 4, 32);
      s += __shfl_xor(s, 8, 32);
      lsum[v] = lsum[v]*corr[v] + s;
      const int row = v + 8*lh;              // C-layout row -> LDS[q][k]
      Pl[row][ll]      = (_Float16)p0;
      Pl[row][16 + ll] = (_Float16)p1;
    }
#pragma unroll
    for (int t2 = 0; t2 < 4; ++t2)
#pragma unroll
      for (int v = 0; v < 8; ++v) O[t2][v] *= corr[v];

    __syncthreads();                          // single-wave wg: DS waits only
    v16h Pa;                                  // re-read as A-fragment (LDS transpose)
    {
      const _Float16* pr = &Pl[ll][0];
      v8h lo = *(const v8h*)(pr + lh*16);
      v8h hi = *(const v8h*)(pr + lh*16 + 8);
#pragma unroll
      for (int i = 0; i < 8; ++i) { Pa[i] = lo[i]; Pa[8+i] = hi[i]; }
    }
    __syncthreads();

    v16h Vb[4];                               // load all V B-frags before WMMA chain
#pragma unroll
    for (int t2 = 0; t2 < 4; ++t2)
      Vb[t2] = *(const v16h*)(vb + (size_t)(t2*16 + ll)*NN + k0 + lh*16);
#pragma unroll
    for (int t2 = 0; t2 < 4; ++t2)
      O[t2] = wmma16(Pa, Vb[t2], O[t2]);
  }

  // emit unnormalized partials: [m(16) | l(16) | O e-major 64x16]
  float* pp = part + ((size_t)sp*1024 + qblk)*PARTSZ;
  if (ll == 0) {
#pragma unroll
    for (int v = 0; v < 8; ++v) {
      pp[v + 8*lh]      = m[v];
      pp[16 + v + 8*lh] = lsum[v];
    }
  }
#pragma unroll
  for (int t2 = 0; t2 < 4; ++t2)
#pragma unroll
    for (int v = 0; v < 8; ++v)
      pp[32 + (t2*16 + ll)*16 + v + 8*lh] = O[t2][v];   // 32B contiguous per lane
}

// ---------------------------------------------------------------------------
// Kernel 2b: combine the NSPLIT partials per query tile -> agg (f16)
// lane: ll = query row in tile, lh = e half (0..31 / 32..63)
// ---------------------------------------------------------------------------
__global__ __launch_bounds__(32) void k_comb(const float* __restrict__ part,
                                             _Float16* __restrict__ aggh) {
  const int qblk = blockIdx.x;                 // 0..1023
  const int b = qblk >> 8, n0 = (qblk & 255) << 4;
  const int l = threadIdx.x, lh = l >> 4, ll = l & 15;

  float ms[NSPLIT];
  float gm = -3e38f;
#pragma unroll
  for (int sp = 0; sp < NSPLIT; ++sp) {
    ms[sp] = part[((size_t)sp*1024 + qblk)*PARTSZ + ll];
    gm = fmaxf(gm, ms[sp]);
  }
  float w[NSPLIT], gl = 0.0f;
#pragma unroll
  for (int sp = 0; sp < NSPLIT; ++sp) {
    w[sp] = __expf(ms[sp] - gm);               // all-masked split: m=-1e30 -> w=0
    gl += w[sp] * part[((size_t)sp*1024 + qblk)*PARTSZ + 16 + ll];
  }

  float acc[32];
#pragma unroll
  for (int i = 0; i < 32; ++i) acc[i] = 0.0f;
#pragma unroll
  for (int sp = 0; sp < NSPLIT; ++sp) {
    const float* po = part + ((size_t)sp*1024 + qblk)*PARTSZ + 32;
#pragma unroll
    for (int e2 = 0; e2 < 32; ++e2)            // e-major: lanes 0..15 read 64B runs
      acc[e2] += w[sp] * po[(lh*32 + e2)*16 + ll];
  }

  const float inv = 1.0f / gl;
  _Float16* ag = aggh + ((size_t)b*NN + n0 + ll)*EE + lh*32;
#pragma unroll
  for (int e2 = 0; e2 < 32; ++e2)
    ag[e2] = (_Float16)(acc[e2] * inv);        // 64B contiguous per lane
}

// ---------------------------------------------------------------------------
// Kernel 3: delta = agg @ Wp^T, residual add at hi queries, coalesced writeback
// ---------------------------------------------------------------------------
__global__ __launch_bounds__(32) void k_out(const float* __restrict__ feat,
    const float* __restrict__ Wp, const _Float16* __restrict__ aggh,
    const unsigned char* __restrict__ maskb, float* __restrict__ out) {
  __shared__ float dbuf[CC][17];               // delta staged as [c][n_local]
  const int blk = blockIdx.x, b = blk >> 8, n0 = (blk & 255) << 4;
  const int l = threadIdx.x, lh = l >> 4, ll = l & 15;

  v16h Aa[2];
  {
    const _Float16* ar = aggh + ((size_t)b*NN + n0 + ll)*EE;
#pragma unroll
    for (int ch = 0; ch < 2; ++ch) {
      v8h lo = *(const v8h*)(ar + ch*32 + lh*16);
      v8h hi = *(const v8h*)(ar + ch*32 + lh*16 + 8);
#pragma unroll
      for (int i = 0; i < 8; ++i) { Aa[ch][i] = lo[i]; Aa[ch][8+i] = hi[i]; }
    }
  }

#pragma unroll
  for (int ct = 0; ct < 8; ++ct) {             // C = 128 -> 8 col tiles
    v16h Bf[2];
#pragma unroll
    for (int ch = 0; ch < 2; ++ch) {           // materialize fragments, then WMMA
      const float* wr = Wp + (size_t)(ct*16 + ll)*EE + ch*32 + lh*16; // Wp[c][e]
#pragma unroll
      for (int i = 0; i < 16; ++i) Bf[ch][i] = (_Float16)wr[i];
    }
    v8f acc = {};
    acc = wmma16(Aa[0], Bf[0], acc);
    acc = wmma16(Aa[1], Bf[1], acc);
#pragma unroll
    for (int v = 0; v < 8; ++v) dbuf[ct*16 + ll][v + 8*lh] = acc[v];
  }
  __syncthreads();

  const bool hi = maskb[(size_t)b*NN + n0 + ll] != 0;
  const float* fb = feat + (size_t)b*CC*NN;
  float* ob = out + (size_t)b*CC*NN;
#pragma unroll 4
  for (int r = 0; r < 64; ++r) {               // 2 channel-rows per pass, 64B segments
    const int c = r*2 + lh;
    const size_t idx = (size_t)c*NN + n0 + ll;
    ob[idx] = fb[idx] + (hi ? dbuf[c][ll] : 0.0f);
  }
}

// ---------------------------------------------------------------------------
extern "C" void kernel_launch(void* const* d_in, const int* in_sizes, int n_in,
                              void* d_out, int out_size, void* d_ws, size_t ws_size,
                              hipStream_t stream) {
  (void)in_sizes; (void)n_in; (void)out_size; (void)ws_size;
  const float* feat = (const float*)d_in[0];
  const float* Wq   = (const float*)d_in[1];
  const float* Wk   = (const float*)d_in[2];
  const float* Wv   = (const float*)d_in[3];
  const float* Wp   = (const float*)d_in[4];
  const unsigned char* mask = (const unsigned char*)d_in[5];  // jax bool -> 1 byte
  float* out = (float*)d_out;

  _Float16* Qh  = (_Float16*)d_ws;                 // each 4*4096*64*2B = 2MB
  _Float16* Kh  = Qh  + (size_t)BB*NN*EE;
  _Float16* Vth = Kh  + (size_t)BB*NN*EE;
  _Float16* Ah  = Vth + (size_t)BB*NN*EE;
  float*    Part = (float*)(Ah + (size_t)BB*NN*EE); // NSPLIT*1024*1056*4B ~= 16.5MB

  const dim3 blockd(32);
  k_proj<<<dim3(BB*(NN/16)),        blockd, 0, stream>>>(feat, Wq, Wk, Wv, Qh, Kh, Vth);
  k_attn<<<dim3(NSPLIT*BB*(NN/16)), blockd, 0, stream>>>(Qh, Kh, Vth, mask, Part);
  k_comb<<<dim3(BB*(NN/16)),        blockd, 0, stream>>>(Part, Ah);
  k_out <<<dim3(BB*(NN/16)),        blockd, 0, stream>>>(feat, Wp, Ah, mask, out);
}
